// Block_12352325943565
// MI455X (gfx1250) — matrix operand order repaired and checked by reading, not compiled, not run on hardware
//
#include <hip/hip_runtime.h>

typedef __attribute__((ext_vector_type(16))) _Float16 v16h;
typedef __attribute__((ext_vector_type(8)))  _Float16 v8h;
typedef __attribute__((ext_vector_type(8)))  float    v8f;

#define E    512
#define NSEQ 4096
#define HDIM 2048
#define NHEAD 8
#define HD   64

__device__ __forceinline__ v16h make_v16(v8h lo, v8h hi) {
  union { v8h h[2]; v16h v; } u;
  u.h[0] = lo; u.h[1] = hi;
  return u.v;
}

// A fragment (16x32 f16, MxK): lane l holds row l%16; K chunks at
// k0+(l/16)*8 .. +7 and k0+16+(l/16)*8 .. +7   (ISA 7.12.2, 16-bit A 16x32)
__device__ __forceinline__ v16h load_fragA(const _Float16* tile, int ld, int k0, int lane) {
  int r    = lane & 15;
  int ksub = (lane >> 4) * 8;
  const _Float16* p = tile + (size_t)r * ld + k0 + ksub;
  v8h lo = *(const v8h*)p;
  v8h hi = *(const v8h*)(p + 16);
  return make_v16(lo, hi);
}

// B fragment (32x16 f16, KxN), sourced from a [N,K] row-major matrix so that
// column n of B == row n of the source (contiguous K). Lane l holds column
// l%16; K chunks at k0+(l/16)*8 .. +7 and k0+16+(l/16)*8 .. +7 (same as A).
__device__ __forceinline__ v16h load_fragB(const _Float16* tile, int ld, int k0, int lane) {
  int n    = lane & 15;
  int ksub = (lane >> 4) * 8;
  const _Float16* p = tile + (size_t)n * ld + k0 + ksub;
  v8h lo = *(const v8h*)p;
  v8h hi = *(const v8h*)(p + 16);
  return make_v16(lo, hi);
}

// Generic batched f16 WMMA GEMM:
//   C[z][M,N] = alpha * A[z][M,K] @ B[z][N,K]^T (+bias)(+relu)(+resid)
// One wave -> 32x64 output tile (2 M-subtiles x 4 N-subtiles -> 8 WMMAs/k-step,
// B fragments reused across both M-subtiles: 12 b128 loads per 8 WMMAs).
// Block = 128 threads = 4 waves -> 128 rows. grid = (M/128, N/64, batch).
// zsA/zsB/zsC are per-z element strides (0 and gridDim.z==1 for unbatched).
__global__ __launch_bounds__(128) void gemm_f16_wmma(
    const _Float16* __restrict__ A, int lda, size_t zsA,
    const _Float16* __restrict__ B, int ldb, size_t zsB,
    const float* __restrict__ bias,
    const float* __restrict__ resid, int ldr,
    float* __restrict__ Cf, int ldc,
    _Float16* __restrict__ Ch, int ldch, size_t zsC,
    int K, float alpha, int relu)
{
  size_t z = blockIdx.z;
  A += z * zsA;
  B += z * zsB;
  if (Cf) Cf += z * zsC;
  if (Ch) Ch += z * zsC;

  int lane = threadIdx.x & 31;
  int wave = threadIdx.x >> 5;
  int m0 = (blockIdx.x * 4 + wave) * 32;
  int n0 = blockIdx.y * 64;
  const _Float16* A0 = A + (size_t)m0 * lda;
  const _Float16* A1 = A + (size_t)(m0 + 16) * lda;
  const _Float16* B0 = B + (size_t)(n0 +  0) * ldb;
  const _Float16* B1 = B + (size_t)(n0 + 16) * ldb;
  const _Float16* B2 = B + (size_t)(n0 + 32) * ldb;
  const _Float16* B3 = B + (size_t)(n0 + 48) * ldb;

  v8f c00 = {}, c01 = {}, c02 = {}, c03 = {};
  v8f c10 = {}, c11 = {}, c12 = {}, c13 = {};
  for (int k = 0; k < K; k += 32) {
    v16h a0 = load_fragA(A0, lda, k, lane);
    v16h a1 = load_fragA(A1, lda, k, lane);
    v16h b0 = load_fragB(B0, ldb, k, lane);
    v16h b1 = load_fragB(B1, ldb, k, lane);
    v16h b2 = load_fragB(B2, ldb, k, lane);
    v16h b3 = load_fragB(B3, ldb, k, lane);
    c00 = __builtin_amdgcn_wmma_f32_16x16x32_f16(false, a0, false, b0, (short)0, c00, false, false);
    c10 = __builtin_amdgcn_wmma_f32_16x16x32_f16(false, a1, false, b0, (short)0, c10, false, false);
    c01 = __builtin_amdgcn_wmma_f32_16x16x32_f16(false, a0, false, b1, (short)0, c01, false, false);
    c11 = __builtin_amdgcn_wmma_f32_16x16x32_f16(false, a1, false, b1, (short)0, c11, false, false);
    c02 = __builtin_amdgcn_wmma_f32_16x16x32_f16(false, a0, false, b2, (short)0, c02, false, false);
    c12 = __builtin_amdgcn_wmma_f32_16x16x32_f16(false, a1, false, b2, (short)0, c12, false, false);
    c03 = __builtin_amdgcn_wmma_f32_16x16x32_f16(false, a0, false, b3, (short)0, c03, false, false);
    c13 = __builtin_amdgcn_wmma_f32_16x16x32_f16(false, a1, false, b3, (short)0, c13, false, false);
  }

  v8f acc[2][4] = {{c00, c01, c02, c03}, {c10, c11, c12, c13}};
  int col    = lane & 15;
  int rowoff = (lane >> 4) * 8;
#pragma unroll
  for (int i = 0; i < 2; ++i) {
#pragma unroll
    for (int j = 0; j < 4; ++j) {
      int n = n0 + j * 16 + col;
      float bv = bias ? bias[n] : 0.0f;
#pragma unroll
      for (int v = 0; v < 8; ++v) {
        int m = m0 + i * 16 + rowoff + v;
        float val = acc[i][j][v] * alpha + bv;
        if (relu)  val = fmaxf(val, 0.0f);
        if (resid) val += resid[(size_t)m * ldr + n];
        if (Cf) Cf[(size_t)m * ldc + n] = val;
        if (Ch) Ch[(size_t)m * ldch + n] = (_Float16)val;
      }
    }
  }
}

// LayerNorm over E=512, one block (256 threads) per row, f16 output.
__global__ __launch_bounds__(256) void layernorm_to_f16(
    const float* __restrict__ X, const float* __restrict__ g,
    const float* __restrict__ b, _Float16* __restrict__ out)
{
  __shared__ float red[256];
  int row = blockIdx.x;
  int t = threadIdx.x;
  const float* x = X + (size_t)row * E;
  float v0 = x[t], v1 = x[t + 256];
  red[t] = v0 + v1;
  __syncthreads();
  for (int s = 128; s > 0; s >>= 1) {
    if (t < s) red[t] += red[t + s];
    __syncthreads();
  }
  float mu = red[0] * (1.0f / E);
  __syncthreads();
  float d0 = v0 - mu, d1 = v1 - mu;
  red[t] = d0 * d0 + d1 * d1;
  __syncthreads();
  for (int s = 128; s > 0; s >>= 1) {
    if (t < s) red[t] += red[t + s];
    __syncthreads();
  }
  float r = rsqrtf(red[0] * (1.0f / E) + 1e-5f);
  _Float16* o = out + (size_t)row * E;
  o[t]       = (_Float16)(d0 * r * g[t] + b[t]);
  o[t + 256] = (_Float16)(d1 * r * g[t + 256] + b[t + 256]);
}

// In-place row softmax on f16 scores, one block (256 threads) per row.
// blockIdx.x = row, blockIdx.y = batch (stride zstride elements).
__global__ __launch_bounds__(256) void softmax_rows(_Float16* __restrict__ S, int N, size_t zstride)
{
  __shared__ float red[256];
  int t = threadIdx.x;
  _Float16* p = S + (size_t)blockIdx.y * zstride + (size_t)blockIdx.x * N;
  float mx = -3.0e38f;
  for (int i = t; i < N; i += 256) mx = fmaxf(mx, (float)p[i]);
  red[t] = mx; __syncthreads();
  for (int s = 128; s > 0; s >>= 1) { if (t < s) red[t] = fmaxf(red[t], red[t + s]); __syncthreads(); }
  mx = red[0];
  __syncthreads();
  float sum = 0.0f;
  for (int i = t; i < N; i += 256) sum += __expf((float)p[i] - mx);
  red[t] = sum; __syncthreads();
  for (int s = 128; s > 0; s >>= 1) { if (t < s) red[t] += red[t + s]; __syncthreads(); }
  float inv = 1.0f / red[0];
  for (int i = t; i < N; i += 256) p[i] = (_Float16)(__expf((float)p[i] - mx) * inv);
}

__global__ void cvt_f32_f16(const float* __restrict__ in, _Float16* __restrict__ out, int n)
{
  int i = blockIdx.x * 256 + threadIdx.x;
  if (i < n) out[i] = (_Float16)in[i];
}

// vT[c, n] = qkv[n, 1024 + c]  for c in [0,512), n in [0,4096)
// (makes B-columns for the PV matmul K-contiguous)
__global__ void transpose_v(const _Float16* __restrict__ qkv, _Float16* __restrict__ vT)
{
  int idx = blockIdx.x * 256 + threadIdx.x;     // over 512*4096
  int n = idx & (NSEQ - 1);
  int c = idx >> 12;
  vT[(size_t)c * NSEQ + n] = qkv[(size_t)n * (3 * E) + 2 * E + c];
}

extern "C" void kernel_launch(void* const* d_in, const int* in_sizes, int n_in,
                              void* d_out, int out_size, void* d_ws, size_t ws_size,
                              hipStream_t stream) {
  const float* x      = (const float*)d_in[0];
  const float* ln_g   = (const float*)d_in[1];
  const float* ln_b   = (const float*)d_in[2];
  const float* qkv_w  = (const float*)d_in[3];
  const float* qkv_b  = (const float*)d_in[4];
  const float* proj_w = (const float*)d_in[5];
  const float* proj_b = (const float*)d_in[6];
  const float* fc1_w  = (const float*)d_in[7];
  const float* fc1_b  = (const float*)d_in[8];
  const float* fc2_w  = (const float*)d_in[9];
  const float* fc2_b  = (const float*)d_in[10];
  const float* fc3_w  = (const float*)d_in[11];
  const float* fc3_b  = (const float*)d_in[12];
  float* out = (float*)d_out;

  const size_t NN = (size_t)NSEQ * NSEQ;            // one head's score matrix

  char* ws = (char*)d_ws;
  auto carve = [&](size_t bytes) -> char* {
    char* p = ws;
    ws += (bytes + 255) & ~(size_t)255;
    return p;
  };
  _Float16* wqkv  = (_Float16*)carve((size_t)3 * E * E * 2);
  _Float16* wproj = (_Float16*)carve((size_t)E * E * 2);
  _Float16* wfc1  = (_Float16*)carve((size_t)HDIM * E * 2);
  _Float16* wfc2  = (_Float16*)carve((size_t)HDIM * HDIM * 2);
  _Float16* wfc3  = (_Float16*)carve((size_t)E * HDIM * 2);
  _Float16* h16   = (_Float16*)carve((size_t)NSEQ * E * 2);       // LN out (reused for LN2)
  _Float16* qkv16 = (_Float16*)carve((size_t)NSEQ * 3 * E * 2);
  _Float16* vT    = (_Float16*)carve((size_t)E * NSEQ * 2);
  _Float16* O16   = (_Float16*)carve((size_t)NSEQ * E * 2);       // attention output
  float*    x1    = (float*)   carve((size_t)NSEQ * E * 4);       // residual 1
  _Float16* h1    = (_Float16*)carve((size_t)NSEQ * HDIM * 2);
  _Float16* h2    = (_Float16*)carve((size_t)NSEQ * HDIM * 2);
  // Score storage last: all heads at once if workspace allows, else one head.
  size_t base_used = (size_t)(ws - (char*)d_ws);
  bool batched = (ws_size >= base_used + (size_t)NHEAD * NN * 2 + 256);
  _Float16* sc = (_Float16*)carve((batched ? (size_t)NHEAD : 1) * NN * 2);

  dim3 blk128(128), blk256(256);
  auto cvt = [&](const float* src, _Float16* dst, int n) {
    cvt_f32_f16<<<dim3((n + 255) / 256), blk256, 0, stream>>>(src, dst, n);
  };
  cvt(qkv_w, wqkv, 3 * E * E);
  cvt(proj_w, wproj, E * E);
  cvt(fc1_w, wfc1, HDIM * E);
  cvt(fc2_w, wfc2, HDIM * HDIM);
  cvt(fc3_w, wfc3, E * HDIM);

  // ---- LN1 ----
  layernorm_to_f16<<<dim3(NSEQ), blk256, 0, stream>>>(x, ln_g, ln_b, h16);

  // ---- qkv = ln1 @ qkv_w^T + qkv_b  -> f16 [4096,1536] ----
  gemm_f16_wmma<<<dim3(NSEQ / 128, (3 * E) / 64), blk128, 0, stream>>>(
      h16, E, 0, wqkv, E, 0, qkv_b, nullptr, 0,
      nullptr, 0, qkv16, 3 * E, 0, E, 1.0f, 0);

  // ---- V^T for PV matmul ----
  transpose_v<<<dim3((E * NSEQ) / 256), blk256, 0, stream>>>(qkv16, vT);

  // ---- attention ----
  const float scale = 0.125f; // 64^-0.5
  if (batched) {
    // All heads via blockIdx.z: full-machine occupancy for scores/softmax/PV.
    gemm_f16_wmma<<<dim3(NSEQ / 128, NSEQ / 64, NHEAD), blk128, 0, stream>>>(
        qkv16, 3 * E, HD,                 // A = Q_h (z-stride 64 cols)
        qkv16 + E, 3 * E, HD,             // B = K_h
        nullptr, nullptr, 0,
        nullptr, 0, sc, NSEQ, NN, HD, scale, 0);
    softmax_rows<<<dim3(NSEQ, NHEAD), blk256, 0, stream>>>(sc, NSEQ, NN);
    gemm_f16_wmma<<<dim3(NSEQ / 128, 1, NHEAD), blk128, 0, stream>>>(
        sc, NSEQ, NN,                     // A = probs_h
        vT, NSEQ, (size_t)HD * NSEQ,      // B = V_h^T
        nullptr, nullptr, 0,
        nullptr, 0, O16, E, HD, NSEQ, 1.0f, 0);
  } else {
    // Per-head streaming through a single score buffer.
    for (int h = 0; h < NHEAD; ++h) {
      gemm_f16_wmma<<<dim3(NSEQ / 128, NSEQ / 64), blk128, 0, stream>>>(
          qkv16 + h * HD, 3 * E, 0, qkv16 + E + h * HD, 3 * E, 0,
          nullptr, nullptr, 0,
          nullptr, 0, sc, NSEQ, 0, HD, scale, 0);
      softmax_rows<<<dim3(NSEQ), blk256, 0, stream>>>(sc, NSEQ, 0);
      gemm_f16_wmma<<<dim3(NSEQ / 128, 1), blk128, 0, stream>>>(
          sc, NSEQ, 0, vT + (size_t)h * HD * NSEQ, NSEQ, 0,
          nullptr, nullptr, 0,
          nullptr, 0, O16 + h * HD, E, 0, NSEQ, 1.0f, 0);
    }
  }

  // ---- x1 = x + O @ proj_w^T + proj_b   (f32) ----
  gemm_f16_wmma<<<dim3(NSEQ / 128, E / 64), blk128, 0, stream>>>(
      O16, E, 0, wproj, E, 0, proj_b, x, E,
      x1, E, nullptr, 0, 0, E, 1.0f, 0);

  // ---- LN2 (same gamma/beta) ----
  layernorm_to_f16<<<dim3(NSEQ), blk256, 0, stream>>>(x1, ln_g, ln_b, h16);

  // ---- MLP ----
  gemm_f16_wmma<<<dim3(NSEQ / 128, HDIM / 64), blk128, 0, stream>>>(
      h16, E, 0, wfc1, E, 0, fc1_b, nullptr, 0,
      nullptr, 0, h1, HDIM, 0, E, 1.0f, 1);
  gemm_f16_wmma<<<dim3(NSEQ / 128, HDIM / 64), blk128, 0, stream>>>(
      h1, HDIM, 0, wfc2, HDIM, 0, fc2_b, nullptr, 0,
      nullptr, 0, h2, HDIM, 0, HDIM, 1.0f, 1);
  gemm_f16_wmma<<<dim3(NSEQ / 128, E / 64), blk128, 0, stream>>>(
      h2, HDIM, 0, wfc3, HDIM, 0, fc3_b, x1, E,
      out, E, nullptr, 0, 0, HDIM, 1.0f, 0);
}
